// Attention_58669253263417
// MI455X (gfx1250) — matrix operand order repaired and checked by reading, not compiled, not run on hardware
//
#include <hip/hip_runtime.h>

typedef __attribute__((ext_vector_type(16))) _Float16 v16h;
typedef __attribute__((ext_vector_type(8)))  float    v8f;

#define HB      64      // BATCH
#define SEQL    512
#define H2D     1024
#define DIN     3072
#define KDIM    1024    // enc part of W1

// ---------------------------------------------------------------------------
// Kernel 1: st_proj[b][h] = b1[h] + sum_{d<2048} st[b,d] * W1[h,d]
// st[b, layer*1024+j] = state[layer][b][j]
// ---------------------------------------------------------------------------
__global__ __launch_bounds__(256) void stproj_kernel(
    const float* __restrict__ state, const float* __restrict__ W1,
    const float* __restrict__ b1, float* __restrict__ out)
{
    int idx = blockIdx.x * 256 + threadIdx.x;   // 0..65535
    int b = idx >> 10;
    int h = idx & 1023;
    float acc = b1[h];
    #pragma unroll
    for (int layer = 0; layer < 2; ++layer) {
        const float4* sr = (const float4*)(state + ((size_t)layer * HB + b) * 1024);
        const float4* wr = (const float4*)(W1 + (size_t)h * DIN + layer * 1024);
        for (int j = 0; j < 256; ++j) {
            float4 s4 = sr[j], w4 = wr[j];
            acc = fmaf(s4.x, w4.x, acc);
            acc = fmaf(s4.y, w4.y, acc);
            acc = fmaf(s4.z, w4.z, acc);
            acc = fmaf(s4.w, w4.w, acc);
        }
    }
    out[idx] = acc;
}

// ---------------------------------------------------------------------------
// Kernel 2: convert W1b (columns 2048..3071 of W1) to f16, row-major [h][k]
// ---------------------------------------------------------------------------
__global__ __launch_bounds__(256) void w1cvt_kernel(
    const float* __restrict__ W1, _Float16* __restrict__ W1h)
{
    int idx = (blockIdx.x * 256 + threadIdx.x) * 4;  // over 1024*1024
    int n = idx >> 10;
    int k = idx & 1023;
    const float* src = W1 + (size_t)n * DIN + 2048 + k;
    _Float16* dst = W1h + idx;
    #pragma unroll
    for (int i = 0; i < 4; ++i) dst[i] = (_Float16)src[i];
}

// ---------------------------------------------------------------------------
// Kernel 3: main WMMA GEMM + fused tanh/dot(W2) epilogue
// Workgroup: 8 waves = 4 (M slabs of 16 rows) x 2 (N slabs of 128 cols)
// M=64 seq rows, N=256 h cols, K=1024 in steps of 32.
// Writes partial scores epart[ntile][b][s].
// ---------------------------------------------------------------------------
__global__ __launch_bounds__(256) void energy_kernel(
    const float* __restrict__ enc,      // (S, B, H2) fp32
    const _Float16* __restrict__ W1h,   // (H2, 1024) f16
    const float* __restrict__ stproj,   // (B, H2) includes b1
    const float* __restrict__ W2,       // (H2)
    float* __restrict__ epart)          // (4, B, S)
{
    int wg = blockIdx.x;          // 0..2047
    int nt = wg & 3;              // N tile (256 cols each)
    int st = (wg >> 2) & 7;       // S tile (64 rows each)
    int b  = wg >> 5;             // batch

    int tid  = threadIdx.x;
    int wave = tid >> 5;          // 0..7
    int lane = tid & 31;
    int wm = wave >> 1;           // 0..3: 16-row slab
    int wn = wave & 1;            // 0..1: 128-col slab
    int g  = lane >> 4;           // lane half
    int ln = lane & 15;

    int srow = st * 64 + wm * 16 + ln;                 // A row (seq index)
    const float* arow = enc + (size_t)srow * (HB * H2D) + (size_t)b * H2D;
    int n0 = nt * 256 + wn * 128;

    v8f acc[8];
    #pragma unroll
    for (int t = 0; t < 8; ++t) acc[t] = (v8f){0.f,0.f,0.f,0.f,0.f,0.f,0.f,0.f};

    for (int k0 = 0; k0 < KDIM; k0 += 32) {
        // A fragment: lane holds row M=ln, K = k0+16g..k0+16g+15
        v16h a;
        {
            const float* p0 = arow + k0 + 16 * g;
            const float* p1 = arow + k0 + 16 * g + 8;
            #pragma unroll
            for (int i = 0; i < 8; ++i) a[i]     = (_Float16)p0[i];
            #pragma unroll
            for (int i = 0; i < 8; ++i) a[8 + i] = (_Float16)p1[i];
        }
        #pragma unroll
        for (int t = 0; t < 8; ++t) {
            // B fragment: lane holds col N = n0+t*16+ln, K = k0+16g..k0+16g+15
            int n = n0 + t * 16 + ln;
            const _Float16* bp = W1h + (size_t)n * KDIM + k0 + 16 * g;
            v16h bb;
            #pragma unroll
            for (int i = 0; i < 16; ++i) bb[i] = bp[i];
            acc[t] = __builtin_amdgcn_wmma_f32_16x16x32_f16(
                false, a, false, bb, (short)0, acc[t], false, false);
        }
    }

    // Epilogue: rowsum[r] = sum_t tanh(acc + stproj[n]) * W2[n]
    // C layout: VGPR r -> row m = r + 8*g, col n = n_tile_base + ln
    float rowsum[8];
    #pragma unroll
    for (int r = 0; r < 8; ++r) rowsum[r] = 0.f;
    #pragma unroll
    for (int t = 0; t < 8; ++t) {
        int n = n0 + t * 16 + ln;
        float sp = stproj[(size_t)b * H2D + n];
        float w2 = W2[n];
        #pragma unroll
        for (int r = 0; r < 8; ++r)
            rowsum[r] += tanhf(acc[t][r] + sp) * w2;
    }

    // Deterministic LDS reduce across 16 lane-columns and 2 wn waves.
    __shared__ float red[8][16][16];
    #pragma unroll
    for (int r = 0; r < 8; ++r)
        red[wave][8 * g + r][ln] = rowsum[r];
    __syncthreads();

    if (tid < 64) {
        int wmr = tid >> 4;       // row slab
        int m   = tid & 15;       // row within slab
        float sum = 0.f;
        #pragma unroll
        for (int wv = 0; wv < 2; ++wv)
            #pragma unroll
            for (int c = 0; c < 16; ++c)
                sum += red[wmr * 2 + wv][m][c];
        epart[((size_t)nt * HB + b) * SEQL + st * 64 + tid] = sum;
    }
}

// ---------------------------------------------------------------------------
// Kernel 4: per-batch softmax over S (deterministic sum of the 4 N-partials)
// ---------------------------------------------------------------------------
__global__ __launch_bounds__(256) void softmax_kernel(
    const float* __restrict__ epart, const float* __restrict__ b2,
    float* __restrict__ alpha)
{
    __shared__ float sc[SEQL];
    __shared__ float red[256];
    int b = blockIdx.x, tid = threadIdx.x;
    float bias = b2[0];
    for (int s = tid; s < SEQL; s += 256) {
        float v = bias;
        #pragma unroll
        for (int nt = 0; nt < 4; ++nt)
            v += epart[((size_t)nt * HB + b) * SEQL + s];
        sc[s] = v;
    }
    __syncthreads();
    red[tid] = fmaxf(sc[tid], sc[tid + 256]);
    __syncthreads();
    for (int off = 128; off > 0; off >>= 1) {
        if (tid < off) red[tid] = fmaxf(red[tid], red[tid + off]);
        __syncthreads();
    }
    float m = red[0];
    __syncthreads();
    float e0 = expf(sc[tid] - m);
    float e1 = expf(sc[tid + 256] - m);
    sc[tid] = e0; sc[tid + 256] = e1;
    red[tid] = e0 + e1;
    __syncthreads();
    for (int off = 128; off > 0; off >>= 1) {
        if (tid < off) red[tid] += red[tid + off];
        __syncthreads();
    }
    float inv = 1.f / red[0];
    alpha[(size_t)b * SEQL + tid]       = sc[tid] * inv;
    alpha[(size_t)b * SEQL + tid + 256] = sc[tid + 256] * inv;
}

// ---------------------------------------------------------------------------
// Kernel 5: context[b][h] = sum_s alpha[b][s] * enc[s][b][h]
// grid = 64 b x 4 h-chunks of 256; coalesced 1KB rows per s.
// ---------------------------------------------------------------------------
__global__ __launch_bounds__(256) void context_kernel(
    const float* __restrict__ enc, const float* __restrict__ alpha,
    float* __restrict__ ctx)
{
    int b  = blockIdx.x & 63;
    int hc = blockIdx.x >> 6;
    int tid = threadIdx.x;
    __shared__ float al[SEQL];
    for (int s = tid; s < SEQL; s += 256) al[s] = alpha[(size_t)b * SEQL + s];
    __syncthreads();
    int h = hc * 256 + tid;
    float acc = 0.f;
    for (int s = 0; s < SEQL; ++s)
        acc = fmaf(al[s], enc[(size_t)s * (HB * H2D) + (size_t)b * H2D + h], acc);
    ctx[(size_t)b * H2D + h] = acc;
}

// ---------------------------------------------------------------------------
extern "C" void kernel_launch(void* const* d_in, const int* in_sizes, int n_in,
                              void* d_out, int out_size, void* d_ws, size_t ws_size,
                              hipStream_t stream)
{
    const float* state = (const float*)d_in[0];   // (2, 64, 1024)
    const float* enc   = (const float*)d_in[1];   // (512, 64, 1024)
    const float* W1    = (const float*)d_in[2];   // (1024, 3072)
    const float* b1    = (const float*)d_in[3];   // (1024)
    const float* W2    = (const float*)d_in[4];   // (1024)
    const float* b2    = (const float*)d_in[5];   // (1)

    float* ctx   = (float*)d_out;                 // context: 64*1024
    float* alpha = ctx + HB * H2D;                // alpha:   64*512

    // workspace layout: stproj (256KB) | epart (512KB) | W1h f16 (2MB)
    float* stp      = (float*)d_ws;
    float* epart    = stp + HB * H2D;
    _Float16* W1h   = (_Float16*)(epart + 4 * HB * SEQL);

    stproj_kernel <<<256,  256, 0, stream>>>(state, W1, b1, stp);
    w1cvt_kernel  <<<1024, 256, 0, stream>>>(W1, W1h);
    energy_kernel <<<2048, 256, 0, stream>>>(enc, W1h, stp, W2, epart);
    softmax_kernel<<<HB,   256, 0, stream>>>(epart, b2, alpha);
    context_kernel<<<256,  256, 0, stream>>>(enc, alpha, ctx);
}
